// TinyAttention_79439715107357
// MI455X (gfx1250) — compile-verified, matrix operand order checked by reading, not run on hardware
//
#include <hip/hip_runtime.h>
#include <hip/hip_bf16.h>

typedef __attribute__((ext_vector_type(16))) _Float16 v16h;
typedef __attribute__((ext_vector_type(8)))  float    v8f;

#define D_MODEL 512
#define D_ATTN  64
#define NSEQ    2048
#define NBATCH  8
#define NTOK    (NBATCH * NSEQ)          // 16384
#define D_OUT   (D_MODEL / 2)            // 256
#define D_QKV   (3 * D_ATTN)             // 192
#define SCALE_F (0.125f)                 // 1/sqrt(64)

static __device__ __forceinline__ v8f wmma_f16(v16h a, v16h b, v8f c) {
    return __builtin_amdgcn_wmma_f32_16x16x32_f16(
        /*neg_a=*/false, a, /*neg_b=*/false, b,
        /*c_mod=*/(short)0, c, /*reuse_a=*/false, /*reuse_b=*/false);
}

// A-fragment (16xK tile, K-slab of 32) from row-major f16 [rows, ld]:
// lane L (0..15): row=L, K 0..7 + 16..23 ; lane L+16: row=L, K 8..15 + 24..31.
// Identical pattern serves the B-fragment when the source is [N, K] row-major.
static __device__ __forceinline__ v16h load_frag_h(const _Float16* __restrict__ base, int ld) {
    const int lane = threadIdx.x & 31;
    const int row  = lane & 15;
    const int kq   = (lane >> 4) << 3;   // 0 or 8
    const _Float16* p = base + (size_t)row * ld + kq;
    v16h a;
#pragma unroll
    for (int i = 0; i < 8; ++i) { a[i] = p[i]; a[i + 8] = p[i + 16]; }
    return a;
}

// Same fragment but from an f32 source, converting in-register.
static __device__ __forceinline__ v16h load_frag_f(const float* __restrict__ base, int ld) {
    const int lane = threadIdx.x & 31;
    const int row  = lane & 15;
    const int kq   = (lane >> 4) << 3;
    const float* p = base + (size_t)row * ld + kq;
    v16h a;
#pragma unroll
    for (int i = 0; i < 8; ++i) { a[i] = (_Float16)p[i]; a[i + 8] = (_Float16)p[i + 16]; }
    return a;
}

// ---------------------------------------------------------------- prep ----
// Transpose W_qkv [512,192] -> [192,512] f16 ; W_out [64,256] -> [256,64] f16.
__global__ void prep_w_kernel(const float* __restrict__ W_qkv,
                              const float* __restrict__ W_out,
                              _Float16* __restrict__ wqkvT,
                              _Float16* __restrict__ woutT) {
    int i = blockIdx.x * blockDim.x + threadIdx.x;
    if (i < D_MODEL * D_QKV) {
        int kk = i / D_QKV, n = i % D_QKV;
        wqkvT[(size_t)n * D_MODEL + kk] = (_Float16)W_qkv[i];
    }
    if (i < D_ATTN * D_OUT) {
        int kk = i / D_OUT, n = i % D_OUT;
        woutT[(size_t)n * D_ATTN + kk] = (_Float16)W_out[i];
    }
}

// ------------------------------------------------------------- QKV GEMM ---
// M=16384, N=192, K=512. One wave per 16x16 tile; q,k stored [tok,64] f16,
// v stored transposed per batch [b][d][n] f16 for contiguous PV B-fragments.
__global__ void qkv_gemm_kernel(const float* __restrict__ x,
                                const _Float16* __restrict__ wqkvT,
                                const float* __restrict__ b_qkv,
                                _Float16* __restrict__ qbuf,
                                _Float16* __restrict__ kbuf,
                                _Float16* __restrict__ vT) {
    const int wave = blockIdx.x * (blockDim.x >> 5) + (threadIdx.x >> 5);
    const int NT = D_QKV / 16;                 // 12
    const int mt = wave / NT, nt = wave % NT;
    if (mt >= NTOK / 16) return;
    const int lane = threadIdx.x & 31;

    v8f c = {};
#pragma unroll 4
    for (int kk = 0; kk < D_MODEL; kk += 32) {
        v16h a = load_frag_f(x + (size_t)mt * 16 * D_MODEL + kk, D_MODEL);
        v16h b = load_frag_h(wqkvT + (size_t)nt * 16 * D_MODEL + kk, D_MODEL);
        c = wmma_f16(a, b, c);
    }

    const int col  = lane & 15;
    const int half = lane >> 4;
    const int ng   = nt * 16 + col;            // global output column 0..191
    const float bias = b_qkv[ng];
    const int dsel = ng >> 6;                  // 0=q 1=k 2=v
    const int d    = ng & 63;
#pragma unroll
    for (int r = 0; r < 8; ++r) {
        const int m = mt * 16 + r + 8 * half;  // token index
        const _Float16 val = (_Float16)(c[r] + bias);
        if (dsel == 0)      qbuf[(size_t)m * D_ATTN + d] = val;
        else if (dsel == 1) kbuf[(size_t)m * D_ATTN + d] = val;
        else {
            const int b_idx = m >> 11, n_in = m & (NSEQ - 1);
            vT[((size_t)b_idx * D_ATTN + d) * NSEQ + n_in] = val;
        }
    }
}

// ------------------------------------------------------- flash attention --
// One wave per (batch, 16-query tile); stream 32 keys/iter with online softmax.
__global__ void attn_kernel(const _Float16* __restrict__ qbuf,
                            const _Float16* __restrict__ kbuf,
                            const _Float16* __restrict__ vT,
                            _Float16* __restrict__ obuf) {
    const int wave = blockIdx.x * (blockDim.x >> 5) + (threadIdx.x >> 5);
    if (wave >= NBATCH * (NSEQ / 16)) return;
    const int b  = wave >> 7;                  // 128 q-tiles per batch
    const int qs = (wave & 127) * 16;
    const int lane = threadIdx.x & 31;
    const int half = lane >> 4;
    const int col  = lane & 15;

    __shared__ _Float16 lds_all[8 * 16 * 32];  // 8 waves * 1KB
    _Float16* lds = lds_all + (threadIdx.x >> 5) * (16 * 32);

    const _Float16* qb = qbuf + ((size_t)b * NSEQ + qs) * D_ATTN;
    const v16h qa = load_frag_h(qb, D_ATTN);        // d 0..31
    const v16h qc = load_frag_h(qb + 32, D_ATTN);   // d 32..63

    float m_r[8], l_r[8];
    v8f o0 = {}, o1 = {}, o2 = {}, o3 = {};
#pragma unroll
    for (int r = 0; r < 8; ++r) { m_r[r] = -1e30f; l_r[r] = 0.f; }

    const int jmax = (qs + 15) >> 5;           // last 32-key tile touching qs+15
    for (int j = 0; j <= jmax; ++j) {
        const int ks = j * 32;
        const _Float16* kb = kbuf + ((size_t)b * NSEQ + ks) * D_ATTN;
        v8f s0 = {}, s1 = {};
        s0 = wmma_f16(qa, load_frag_h(kb, D_ATTN), s0);
        s0 = wmma_f16(qc, load_frag_h(kb + 32, D_ATTN), s0);
        s1 = wmma_f16(qa, load_frag_h(kb + 16 * D_ATTN, D_ATTN), s1);
        s1 = wmma_f16(qc, load_frag_h(kb + 16 * D_ATTN + 32, D_ATTN), s1);

        // scale + causal mask
#pragma unroll
        for (int r = 0; r < 8; ++r) {
            const int qi = qs + r + 8 * half;
            float a0 = s0[r] * SCALE_F; if (ks + col > qi)      a0 = -1e30f;
            float a1 = s1[r] * SCALE_F; if (ks + 16 + col > qi) a1 = -1e30f;
            s0[r] = a0; s1[r] = a1;
        }
        // online softmax: per-row max/sum via xor-shuffle within 16-lane half
#pragma unroll
        for (int r = 0; r < 8; ++r) {
            float mx = fmaxf(s0[r], s1[r]);
#pragma unroll
            for (int off = 8; off >= 1; off >>= 1) mx = fmaxf(mx, __shfl_xor(mx, off, 32));
            const float mn = fmaxf(m_r[r], mx);
            const float p0 = __expf(s0[r] - mn);
            const float p1 = __expf(s1[r] - mn);
            const int row = r + 8 * half;
            lds[row * 32 + col]      = (_Float16)p0;
            lds[row * 32 + 16 + col] = (_Float16)p1;
            float sum = p0 + p1;
#pragma unroll
            for (int off = 8; off >= 1; off >>= 1) sum += __shfl_xor(sum, off, 32);
            const float corr = __expf(m_r[r] - mn);
            l_r[r] = l_r[r] * corr + sum;
            m_r[r] = mn;
            o0[r] *= corr; o1[r] *= corr; o2[r] *= corr; o3[r] *= corr;
        }
        // reload P as an A-fragment (LDS bounce does the layout change)
        v16h pf;
        {
            const int row = lane & 15;
            const int kq  = (lane >> 4) << 3;
#pragma unroll
            for (int i = 0; i < 8; ++i) {
                pf[i]     = lds[row * 32 + kq + i];
                pf[i + 8] = lds[row * 32 + kq + 16 + i];
            }
        }
        // O += P @ V ; V^T[b][d][n] rows are contiguous in keys
        const _Float16* vb = vT + (size_t)b * D_ATTN * NSEQ + ks;
        o0 = wmma_f16(pf, load_frag_h(vb +  0 * NSEQ, NSEQ), o0);
        o1 = wmma_f16(pf, load_frag_h(vb + 16 * NSEQ, NSEQ), o1);
        o2 = wmma_f16(pf, load_frag_h(vb + 32 * NSEQ, NSEQ), o2);
        o3 = wmma_f16(pf, load_frag_h(vb + 48 * NSEQ, NSEQ), o3);
    }

#pragma unroll
    for (int r = 0; r < 8; ++r) {
        const float inv = 1.f / l_r[r];
        const int tok = qs + r + 8 * half;
        _Float16* op = obuf + ((size_t)b * NSEQ + tok) * D_ATTN;
        op[col]      = (_Float16)(o0[r] * inv);
        op[16 + col] = (_Float16)(o1[r] * inv);
        op[32 + col] = (_Float16)(o2[r] * inv);
        op[48 + col] = (_Float16)(o3[r] * inv);
    }
}

// ------------------------------------------------------------ out GEMM ----
// M=16384, N=256, K=64 ; f32 output + bias.
__global__ void out_gemm_kernel(const _Float16* __restrict__ obuf,
                                const _Float16* __restrict__ woutT,
                                const float* __restrict__ b_out,
                                float* __restrict__ out) {
    const int wave = blockIdx.x * (blockDim.x >> 5) + (threadIdx.x >> 5);
    const int mt = wave >> 4, nt = wave & 15;
    if (mt >= NTOK / 16) return;
    const int lane = threadIdx.x & 31;

    v8f c = {};
    c = wmma_f16(load_frag_h(obuf + (size_t)mt * 16 * D_ATTN, D_ATTN),
                 load_frag_h(woutT + (size_t)nt * 16 * D_ATTN, D_ATTN), c);
    c = wmma_f16(load_frag_h(obuf + (size_t)mt * 16 * D_ATTN + 32, D_ATTN),
                 load_frag_h(woutT + (size_t)nt * 16 * D_ATTN + 32, D_ATTN), c);

    const int col = lane & 15, half = lane >> 4;
    const float bias = b_out[nt * 16 + col];
#pragma unroll
    for (int r = 0; r < 8; ++r) {
        const int m = mt * 16 + r + 8 * half;
        out[(size_t)m * D_OUT + nt * 16 + col] = c[r] + bias;
    }
}

extern "C" void kernel_launch(void* const* d_in, const int* in_sizes, int n_in,
                              void* d_out, int out_size, void* d_ws, size_t ws_size,
                              hipStream_t stream) {
    (void)in_sizes; (void)n_in; (void)out_size; (void)ws_size;
    const float* x     = (const float*)d_in[0];
    const float* W_qkv = (const float*)d_in[1];
    const float* b_qkv = (const float*)d_in[2];
    const float* W_out = (const float*)d_in[3];
    const float* b_out = (const float*)d_in[4];
    float* out = (float*)d_out;

    char* ws = (char*)d_ws;
    _Float16* wqkvT = (_Float16*)ws; ws += (size_t)D_QKV * D_MODEL * 2;   // 192K
    _Float16* woutT = (_Float16*)ws; ws += (size_t)D_OUT * D_ATTN * 2;    // 32K
    _Float16* qbuf  = (_Float16*)ws; ws += (size_t)NTOK * D_ATTN * 2;     // 2M
    _Float16* kbuf  = (_Float16*)ws; ws += (size_t)NTOK * D_ATTN * 2;     // 2M
    _Float16* vT    = (_Float16*)ws; ws += (size_t)NTOK * D_ATTN * 2;     // 2M
    _Float16* obuf  = (_Float16*)ws; ws += (size_t)NTOK * D_ATTN * 2;     // 2M

    // prep: 512*192 = 98304 elements dominates
    prep_w_kernel<<<(D_MODEL * D_QKV + 255) / 256, 256, 0, stream>>>(
        W_qkv, W_out, wqkvT, woutT);

    // qkv: 1024 m-tiles * 12 n-tiles = 12288 waves / 8 per block
    qkv_gemm_kernel<<<12288 / 8, 256, 0, stream>>>(x, wqkvT, b_qkv, qbuf, kbuf, vT);

    // attention: 8 batches * 128 q-tiles = 1024 waves / 8 per block
    attn_kernel<<<1024 / 8, 256, 0, stream>>>(qbuf, kbuf, vT, obuf);

    // out proj: 1024 m-tiles * 16 n-tiles = 16384 waves / 8 per block
    out_gemm_kernel<<<16384 / 8, 256, 0, stream>>>(obuf, woutT, b_out, out);
}
